// VariableSelectionNetwork_19061064859718
// MI455X (gfx1250) — matrix operand order repaired and checked by reading, not compiled, not run on hardware
//
#include <hip/hip_runtime.h>

// MI455X / gfx1250 implementation.
//
// Roofline: dominant work is 32 per-feature GEMMs. After fusing fc2+GLU-proj
// (no nonlinearity between them) into Wcomb[f] = fw2[f] @ fglu_w[f] (128x256),
// compute is 32 * (16384 x 128 x 256) MACs ~= 34 GFLOP -> matrix-core bound
// (bf16 WMMA, f32 accumulate). Min HBM bytes ~20MB (~1us @ 23.3TB/s); Wcomb
// (2MB bf16) is L2-resident and staged into LDS by the Tensor Data Mover
// (tensor_load_to_lds, TENSORcnt) so no wave spends VGPRs/issue slots copying.

#define T_TOK 16384
#define FDIM  32
#define HDIM  128
#define TWOH  256
#define MT    64          // tokens per workgroup in the main kernel

typedef __bf16 bf16;
typedef __bf16 v16bf __attribute__((ext_vector_type(16)));
typedef __bf16 v8bf  __attribute__((ext_vector_type(8)));
typedef float  v8f   __attribute__((ext_vector_type(8)));
typedef unsigned int u32x4 __attribute__((ext_vector_type(4)));
typedef int          i32x4 __attribute__((ext_vector_type(4)));
typedef int          i32x8 __attribute__((ext_vector_type(8)));

__device__ __forceinline__ float fast_sigmoid(float v) { return 1.f / (1.f + __expf(-v)); }
__device__ __forceinline__ float fast_elu(float v)     { return v > 0.f ? v : (__expf(v) - 1.f); }

// ---------------------------------------------------------------------------
// TDM: DMA one contiguous 32KB block (4096 x 8B) from global into LDS.
// D# built per CDNA5 ISA ch.8: group0 = {count=1, lds_addr, global_addr(57b),
// type=2}; group1 = {data_size=8B, tensor_dim0=tile_dim0=4096, stride=4096}.
// Issued by a single wave; completion tracked with TENSORcnt.
// ---------------------------------------------------------------------------
__device__ __forceinline__ void tdm_load_32kb(const void* gsrc, void* ldst) {
    const unsigned long long ga = (unsigned long long)(uintptr_t)gsrc;
    const unsigned int       la = (unsigned int)(uintptr_t)ldst;  // low 32b = LDS offset

    u32x4 g0;
    g0[0] = 1u;                                            // count=1 (valid user D#)
    g0[1] = la;                                            // lds_addr [63:32]
    g0[2] = (unsigned int)ga;                              // global_addr [95:64]
    g0[3] = (unsigned int)((ga >> 32) & 0x01FFFFFFu)       // global_addr [120:96]
            | (2u << 30);                                  // type=2 ("image") [127:126]

    i32x8 g1;
    g1[0] = (int)(3u << 16);        // workgroup_mask=0, data_size=3 (8 bytes)
    g1[1] = (int)(0x1000u << 16);   // tensor_dim0[15:0]=4096 in bits [63:48]
    g1[2] = (int)(1u << 16);        // tensor_dim0[31:16]=0, tensor_dim1[15:0]=1
    g1[3] = (int)(0x1000u << 16);   // tensor_dim1 hi=0, tile_dim0=4096
    g1[4] = 0;                      // tile_dim1=0, tile_dim2=0 (1-D tile)
    g1[5] = 4096;                   // tensor_dim0_stride[31:0]
    g1[6] = 0;                      // stride hi, tensor_dim1_stride lo
    g1[7] = 0;

    i32x4 z4; z4[0] = 0; z4[1] = 0; z4[2] = 0; z4[3] = 0;
#if defined(__clang_major__) && (__clang_major__ >= 23)
    i32x8 z8; z8[0]=0; z8[1]=0; z8[2]=0; z8[3]=0; z8[4]=0; z8[5]=0; z8[6]=0; z8[7]=0;
    __builtin_amdgcn_tensor_load_to_lds(g0, g1, z4, z4, z8, 0);
#else
    __builtin_amdgcn_tensor_load_to_lds(g0, g1, z4, z4, 0);
#endif
}

// ---------------------------------------------------------------------------
// Kernel 1: WcombT[f][n][k] = sum_o fw2[f][k][o] * fglu_w[f][o][n]  (bf16 out)
// Stored N-major so a WMMA B-fragment lane reads 16 contiguous K values.
// ---------------------------------------------------------------------------
__global__ void vsn_combine_w(const float* __restrict__ fw2,
                              const float* __restrict__ fglu_w,
                              bf16* __restrict__ wcombT) {
    __shared__ float As[16][17];   // [k_local][o_local]
    __shared__ float Bs[16][17];   // [o_local][n_local]
    const int f  = blockIdx.z;
    const int nt = blockIdx.y;     // 16 tiles of N (256)
    const int kt = blockIdx.x;     // 8 tiles of K (128)
    const int tx = threadIdx.x, ty = threadIdx.y;
    const float* fw2f = fw2 + (size_t)f * HDIM * HDIM;
    const float* fgf  = fglu_w + (size_t)f * HDIM * TWOH;
    float acc = 0.f;
    for (int oc = 0; oc < HDIM / 16; ++oc) {
        As[ty][tx] = fw2f[(kt * 16 + ty) * HDIM + oc * 16 + tx];
        Bs[ty][tx] = fgf[(oc * 16 + ty) * TWOH + nt * 16 + tx];
        __syncthreads();
#pragma unroll
        for (int o = 0; o < 16; ++o)
            acc += As[tx][o] * Bs[o][ty];   // out k_local = tx, n_local = ty
        __syncthreads();
    }
    wcombT[(size_t)f * TWOH * HDIM + (nt * 16 + ty) * HDIM + kt * 16 + tx] = (bf16)acc;
}

// ---------------------------------------------------------------------------
// Kernel 2: bcomb[f][n] = sum_o fb2[f][o] * fglu_w[f][o][n] + fglu_b[f][n]
// ---------------------------------------------------------------------------
__global__ void vsn_bcomb(const float* __restrict__ fb2,
                          const float* __restrict__ fglu_w,
                          const float* __restrict__ fglu_b,
                          float* __restrict__ bcomb) {
    const int f = blockIdx.x;
    const int n = threadIdx.x;   // 256 threads
    const float* fgf = fglu_w + (size_t)f * HDIM * TWOH;
    const float* fb  = fb2 + f * HDIM;
    float acc = fglu_b[f * TWOH + n];
    for (int o = 0; o < HDIM; ++o) acc += fb[o] * fgf[o * TWOH + n];
    bcomb[f * TWOH + n] = acc;
}

// ---------------------------------------------------------------------------
// Kernel 3: weight GRN + softmax -> selection weights (T x 32).
// One 128-thread block per token; tiny FLOPs, scalar f32, wave32 shuffles.
// ---------------------------------------------------------------------------
__global__ __launch_bounds__(128) void vsn_weights(
        const float* __restrict__ x,
        const float* __restrict__ ww1, const float* __restrict__ wb1,
        const float* __restrict__ ww2, const float* __restrict__ wb2,
        const float* __restrict__ wglu_w, const float* __restrict__ wglu_b,
        const float* __restrict__ wln_g, const float* __restrict__ wln_b,
        float* __restrict__ wout) {
    __shared__ float xl[FDIM];
    __shared__ float whl[HDIM];
    __shared__ float wfl[FDIM];
    __shared__ float wgl[2 * FDIM];
    const int tok = blockIdx.x;
    const int t   = threadIdx.x;
    if (t < FDIM) xl[t] = x[tok * FDIM + t];
    __syncthreads();
    // fc1 (F->H) + ELU
    float a = wb1[t];
#pragma unroll
    for (int fi = 0; fi < FDIM; ++fi) a += xl[fi] * ww1[fi * HDIM + t];
    whl[t] = fast_elu(a);
    __syncthreads();
    // fc2 (H->F)
    if (t < FDIM) {
        float s = wb2[t];
        for (int h = 0; h < HDIM; ++h) s += whl[h] * ww2[h * FDIM + t];
        wfl[t] = s;
    }
    __syncthreads();
    // GLU projection (F->2F)
    if (t < 2 * FDIM) {
        float s = wglu_b[t];
#pragma unroll
        for (int fi = 0; fi < FDIM; ++fi) s += wfl[fi] * wglu_w[fi * 2 * FDIM + t];
        wgl[t] = s;
    }
    __syncthreads();
    // GLU + identity skip + LayerNorm(F) + softmax — lives entirely in wave 0.
    if (t < FDIM) {
        float pre = wgl[t] * fast_sigmoid(wgl[FDIM + t]) + xl[t];
        float s = pre, ss = pre * pre;
#pragma unroll
        for (int m = 16; m >= 1; m >>= 1) {
            s  += __shfl_xor(s,  m, 32);
            ss += __shfl_xor(ss, m, 32);
        }
        float mean = s * (1.f / FDIM);
        float var  = ss * (1.f / FDIM) - mean * mean;
        float w = (pre - mean) * rsqrtf(var + 1e-5f) * wln_g[t] + wln_b[t];
        float mx = w;
#pragma unroll
        for (int m = 16; m >= 1; m >>= 1) mx = fmaxf(mx, __shfl_xor(mx, m, 32));
        float e = __expf(w - mx);
        float se = e;
#pragma unroll
        for (int m = 16; m >= 1; m >>= 1) se += __shfl_xor(se, m, 32);
        wout[tok * FDIM + t] = e / se;
    }
}

// ---------------------------------------------------------------------------
// Kernel 4: fused per-feature GRNs + weighted combine.
// WG = 256 threads (8 wave32), MT=64 tokens. Wave w -> M-subtile (w&3, 16
// tokens) x paired v/gate column half (w>>2, 64 cols each). Per feature:
//   A (64x128 bf16)  = elu(x_f * fw1 + fb1) staged in LDS
//   B half (128x128) = WcombT[f] DMA'd into LDS by the TDM (wave 0 issues,
//                      s_wait_tensorcnt + barrier for visibility)
//   32 v_wmma_f32_16x16x32_bf16 per wave per feature, f32 accumulators
//   epilogue: GLU + skip + LayerNorm(H=128, cross-wave via LDS partials),
//             out += weight[tok][f] * processed  (kept in registers)
// ---------------------------------------------------------------------------
__global__ __launch_bounds__(256) void vsn_main(
        const float* __restrict__ x,
        const float* __restrict__ fw1, const float* __restrict__ fb1,
        const float* __restrict__ fskip,
        const float* __restrict__ flng, const float* __restrict__ flnb,
        const bf16*  __restrict__ wcombT, const float* __restrict__ bcomb,
        const float* __restrict__ wsel,   // selection weights (T x F)
        float* __restrict__ out) {
    __shared__ alignas(32) bf16 Al[MT * HDIM];      // 16KB, A tile [m][k]
    __shared__ alignas(32) bf16 Bl[HDIM * HDIM];    // 32KB, one B half [n][k]
    __shared__ float xl[MT * FDIM];                 // 8KB,  x tile
    __shared__ float ps[MT * 2], pq[MT * 2];        // LN partials [row][col-half]

    const int t    = threadIdx.x;
    const int wave = t >> 5, lane = t & 31;
    const int msub = wave & 3;        // which 16-token subtile
    const int qh   = wave >> 2;       // which 64-column v/gate pair
    const int hg   = lane >> 4;       // half-group (ISA lane split)
    const int c    = lane & 15;       // column within 16x16 tile / M within A
    const int tok0 = blockIdx.x * MT;
    const int rbase = msub * 16 + hg * 8;   // first of this lane's 8 C-rows

    // Stage x tile once (64 tokens x 32 features).
#pragma unroll
    for (int i = 0; i < 8; ++i) xl[t + i * 256] = x[tok0 * FDIM + t + i * 256];

    v8f accO[4];
#pragma unroll
    for (int nt = 0; nt < 4; ++nt)
#pragma unroll
        for (int j = 0; j < 8; ++j) accO[nt][j] = 0.f;

    for (int f = 0; f < FDIM; ++f) {
        __syncthreads();   // Al/Bl/ps/pq free for reuse
        // ---- stage A = bf16(elu(x_f * fw1 + fb1)), 32 elems/thread ----
        {
            const int m  = t >> 2;
            const int kb = (t & 3) * 32;
            const float xv = xl[m * FDIM + f];
            const float* w1 = fw1 + f * HDIM + kb;
            const float* b1 = fb1 + f * HDIM + kb;
#pragma unroll
            for (int i = 0; i < 32; ++i)
                Al[m * HDIM + kb + i] = (bf16)fast_elu(xv * w1[i] + b1[i]);
        }
        // per-lane row data for this feature
        float wrow[8], xrow[8];
#pragma unroll
        for (int j = 0; j < 8; ++j) {
            const int r = rbase + j;
            xrow[j] = xl[r * FDIM + f];
            wrow[j] = wsel[(size_t)(tok0 + r) * FDIM + f];
        }
        // prefetch next feature's combined weights toward L2 for the TDM
        if (f + 1 < FDIM)
            __builtin_prefetch(wcombT + (size_t)(f + 1) * TWOH * HDIM + t * 128, 0, 1);

        v8f acc[2][4];
        for (int p = 0; p < 2; ++p) {            // p=0: v half, p=1: gate half
            __syncthreads();                     // A ready / prior B reads done
            if (wave == 0) {
                // DMA the 32KB B half into LDS via the Tensor Data Mover.
                tdm_load_32kb(wcombT + (size_t)f * TWOH * HDIM
                              + (size_t)p * HDIM * HDIM, Bl);
                __builtin_amdgcn_s_wait_tensorcnt((short)0);
            }
            __syncthreads();                     // B in LDS visible to all waves
            // init accumulators from combined bias (same value down a column)
#pragma unroll
            for (int nt = 0; nt < 4; ++nt) {
                const float bv = bcomb[f * TWOH + p * HDIM + qh * 64 + nt * 16 + c];
#pragma unroll
                for (int j = 0; j < 8; ++j) acc[p][nt][j] = bv;
            }
#pragma unroll
            for (int ks = 0; ks < 4; ++ks) {     // K = 128 in steps of 32
                // A fragment: lane m=c, group hg holds 16 contiguous K
                const bf16* arow = Al + (msub * 16 + c) * HDIM + ks * 32;
                const v8bf a0 = *(const v8bf*)(arow + hg * 16);       // K 0..7  (+16g)
                const v8bf a1 = *(const v8bf*)(arow + hg * 16 + 8);   // K 8..15 (+16g)
                const v16bf av = __builtin_shufflevector(a0, a1,
                        0, 1, 2, 3, 4, 5, 6, 7, 8, 9, 10, 11, 12, 13, 14, 15);
#pragma unroll
                for (int nt = 0; nt < 4; ++nt) {
                    // B fragment: lane n=c, group hg holds 16 contiguous K
                    const bf16* brow = Bl + (qh * 64 + nt * 16 + c) * HDIM
                                       + ks * 32 + hg * 16;
                    const v16bf bv = *(const v16bf*)brow;
                    acc[p][nt] = __builtin_amdgcn_wmma_f32_16x16x32_bf16(
                            false, av, false, bv, (short)0, acc[p][nt], false, false);
                }
            }
        }

        // ---- epilogue: GLU + skip, LayerNorm over H=128, weighted combine ----
        float fsA[4], gA[4], bA[4];
#pragma unroll
        for (int nt = 0; nt < 4; ++nt) {
            const int h = qh * 64 + nt * 16 + c;
            fsA[nt] = fskip[f * HDIM + h];
            gA[nt]  = flng[f * HDIM + h];
            bA[nt]  = flnb[f * HDIM + h];
        }
        float s[8], ss[8];
#pragma unroll
        for (int j = 0; j < 8; ++j) { s[j] = 0.f; ss[j] = 0.f; }
#pragma unroll
        for (int nt = 0; nt < 4; ++nt)
#pragma unroll
            for (int j = 0; j < 8; ++j) {
                const float glu = acc[0][nt][j] * fast_sigmoid(acc[1][nt][j])
                                  + xrow[j] * fsA[nt];
                acc[0][nt][j] = glu;      // keep for normalization
                s[j]  += glu;
                ss[j] += glu * glu;
            }
        // reduce across the 16 lanes of each half-group (cols of this wave)
#pragma unroll
        for (int j = 0; j < 8; ++j) {
#pragma unroll
            for (int m = 8; m >= 1; m >>= 1) {
                s[j]  += __shfl_xor(s[j],  m, 32);
                ss[j] += __shfl_xor(ss[j], m, 32);
            }
        }
        if (c == 0) {
#pragma unroll
            for (int j = 0; j < 8; ++j) {
                ps[(rbase + j) * 2 + qh] = s[j];
                pq[(rbase + j) * 2 + qh] = ss[j];
            }
        }
        __syncthreads();   // combine the two column-half waves
        float mu[8], rs[8];
#pragma unroll
        for (int j = 0; j < 8; ++j) {
            const int r = rbase + j;
            const float m_ = (ps[r * 2] + ps[r * 2 + 1]) * (1.f / HDIM);
            const float v_ = (pq[r * 2] + pq[r * 2 + 1]) * (1.f / HDIM) - m_ * m_;
            mu[j] = m_;
            rs[j] = rsqrtf(v_ + 1e-5f);
        }
#pragma unroll
        for (int nt = 0; nt < 4; ++nt)
#pragma unroll
            for (int j = 0; j < 8; ++j) {
                const float pr = (acc[0][nt][j] - mu[j]) * rs[j] * gA[nt] + bA[nt];
                accO[nt][j] += wrow[j] * pr;
            }
    }

    // ---- store out (T x 128): lanes 0-15 write 16 contiguous floats ----
#pragma unroll
    for (int nt = 0; nt < 4; ++nt) {
        const int h = qh * 64 + nt * 16 + c;
#pragma unroll
        for (int j = 0; j < 8; ++j)
            out[(size_t)(tok0 + rbase + j) * HDIM + h] = accO[nt][j];
    }
}

// ---------------------------------------------------------------------------
extern "C" void kernel_launch(void* const* d_in, const int* in_sizes, int n_in,
                              void* d_out, int out_size, void* d_ws, size_t ws_size,
                              hipStream_t stream) {
    const float* x      = (const float*)d_in[0];
    const float* fw1    = (const float*)d_in[1];
    const float* fb1    = (const float*)d_in[2];
    const float* fw2    = (const float*)d_in[3];
    const float* fb2    = (const float*)d_in[4];
    const float* fglu_w = (const float*)d_in[5];
    const float* fglu_b = (const float*)d_in[6];
    const float* fskip  = (const float*)d_in[7];
    const float* fln_g  = (const float*)d_in[8];
    const float* fln_b  = (const float*)d_in[9];
    const float* ww1    = (const float*)d_in[10];
    const float* wb1    = (const float*)d_in[11];
    const float* ww2    = (const float*)d_in[12];
    const float* wb2    = (const float*)d_in[13];
    const float* wglu_w = (const float*)d_in[14];
    const float* wglu_b = (const float*)d_in[15];
    const float* wln_g  = (const float*)d_in[16];
    const float* wln_b  = (const float*)d_in[17];

    float* out  = (float*)d_out;                      // (B,S,H) flat
    float* wout = out + (size_t)T_TOK * HDIM;         // (B,S,F) flat

    bf16*  wcombT = (bf16*)d_ws;                      // 32*256*128 bf16 = 2MB
    float* bcomb  = (float*)((char*)d_ws + (size_t)FDIM * TWOH * HDIM * sizeof(bf16));

    // 1) fold fc2 into GLU projection: Wcomb = fw2 @ fglu_w (bf16, N-major)
    vsn_combine_w<<<dim3(8, 16, 32), dim3(16, 16), 0, stream>>>(fw2, fglu_w, wcombT);
    // 2) combined bias
    vsn_bcomb<<<FDIM, TWOH, 0, stream>>>(fb2, fglu_w, fglu_b, bcomb);
    // 3) selection weights (also final output component #2)
    vsn_weights<<<T_TOK, 128, 0, stream>>>(x, ww1, wb1, ww2, wb2,
                                           wglu_w, wglu_b, wln_g, wln_b, wout);
    // 4) fused per-feature GRNs + weighted combine (WMMA core)
    vsn_main<<<T_TOK / MT, 256, 0, stream>>>(x, fw1, fb1, fskip, fln_g, fln_b,
                                             wcombT, bcomb, wout, out);
}
